// GuidedMultiHeadAttention_6640019440204
// MI455X (gfx1250) — matrix operand order repaired and checked by reading, not compiled, not run on hardware
//
#include <hip/hip_runtime.h>
#include <hip/hip_bf16.h>
#include <math.h>

#define B_     8
#define NP1    785
#define DIM_   768
#define H_     12
#define DH     64
#define MROWS  (B_*NP1)      // 6280
#define NPAD   800           // keys padded to 50 tiles of 16
#define ALPHA_ 10.0f
#define EPS_   1e-6f

typedef __attribute__((ext_vector_type(16))) __bf16 bf16x16;
typedef __attribute__((ext_vector_type(8)))  __bf16 bf16x8;
typedef __attribute__((ext_vector_type(8)))  float  v8f;

// ---- WMMA fragment loaders (bf16, 16x16x32, layouts per CDNA5 ISA) ----
// A (16x32): lane = {kh[0], m[3:0]}; VGPR0-3 hold K = kh*8+0..7, VGPR4-7 hold K = 16+kh*8+0..7
static __device__ __forceinline__ bf16x16
load_a_frag(const __bf16* base, int ld, int row0, int maxrow, int k0, int lane) {
  int m  = lane & 15;
  int kh = lane >> 4;
  int r  = row0 + m; if (r > maxrow) r = maxrow;
  const __bf16* p = base + (size_t)r * ld + k0 + kh * 8;
  union { bf16x16 v; bf16x8 h[2]; } u;
  u.h[0] = *(const bf16x8*)(p);        // K = k0+kh*8 .. +7
  u.h[1] = *(const bf16x8*)(p + 16);   // K = k0+16+kh*8 .. +7
  return u.v;
}

// B (32x16): lane = {kh[0], n[3:0]}; VGPR0-3 hold K = kh*8+0..7, VGPR4-7 hold K = 16+kh*8+0..7
static __device__ __forceinline__ bf16x16
load_b_frag(const __bf16* base, int ld, int col0, int maxcol, int k0, int lane) {
  int n  = lane & 15;
  int kh = lane >> 4;
  int c  = col0 + n; if (c > maxcol) c = maxcol;
  const __bf16* p = base + (size_t)c * ld + k0 + kh * 8;
  union { bf16x16 v; bf16x8 h[2]; } u;
  u.h[0] = *(const bf16x8*)(p);
  u.h[1] = *(const bf16x8*)(p + 16);
  return u.v;
}

static __device__ __forceinline__ v8f
wmma_bf16(bf16x16 a, bf16x16 b, v8f c) {
  return __builtin_amdgcn_wmma_f32_16x16x32_bf16(false, a, false, b, (short)0, c, false, false);
}

// ---- fp32 -> bf16 conversion (grid-stride) ----
__global__ void cvt_f32_bf16(const float* __restrict__ src, __bf16* __restrict__ dst, int n) {
  int i = blockIdx.x * blockDim.x + threadIdx.x;
  int stride = gridDim.x * blockDim.x;
  for (; i < n; i += stride) dst[i] = (__bf16)src[i];
}

// ---- projection GEMM: Y = A(bf16, MROWSxDIM) * W(bf16, DIMxDIM)^T + bias ----
// One wave computes a 16x64 strip; ping-pong double buffering (no rotation copies).
// mode 0: bf16 row-major [MROWS x DIM]      (Q, K)
// mode 1: bf16 transposed Vt[b][h][d][NPAD] (V)
// mode 2: fp32 row-major to d_out           (final output)
__global__ __launch_bounds__(32)
void proj_gemm(const __bf16* __restrict__ A, const __bf16* __restrict__ W,
               const float* __restrict__ bias, void* __restrict__ out, int mode) {
  const int KT = DIM_ / 32;               // 24 (even)
  int lane = threadIdx.x;
  int col0 = blockIdx.x * 64;
  int row0 = blockIdx.y * 16;
  v8f acc[4] = {{}, {}, {}, {}};

  bf16x16 aA, bA[4], aB, bB[4];
  aA = load_a_frag(A, DIM_, row0, MROWS - 1, 0, lane);
  #pragma unroll
  for (int j = 0; j < 4; ++j)
    bA[j] = load_b_frag(W, DIM_, col0 + j * 16, DIM_ - 1, 0, lane);

  for (int kt = 0; kt < KT; kt += 2) {
    int k1 = (kt + 1) * 32;
    aB = load_a_frag(A, DIM_, row0, MROWS - 1, k1, lane);
    #pragma unroll
    for (int j = 0; j < 4; ++j)
      bB[j] = load_b_frag(W, DIM_, col0 + j * 16, DIM_ - 1, k1, lane);
    #pragma unroll
    for (int j = 0; j < 4; ++j)
      acc[j] = wmma_bf16(aA, bA[j], acc[j]);
    if (kt + 2 < KT) {
      int k2 = (kt + 2) * 32;
      aA = load_a_frag(A, DIM_, row0, MROWS - 1, k2, lane);
      #pragma unroll
      for (int j = 0; j < 4; ++j)
        bA[j] = load_b_frag(W, DIM_, col0 + j * 16, DIM_ - 1, k2, lane);
    }
    #pragma unroll
    for (int j = 0; j < 4; ++j)
      acc[j] = wmma_bf16(aB, bB[j], acc[j]);
  }

  int n = lane & 15;
  int half = lane >> 4;
  #pragma unroll
  for (int j = 0; j < 4; ++j) {
    int col = col0 + j * 16 + n;
    float bv = bias[col];
    #pragma unroll
    for (int i = 0; i < 8; ++i) {
      int row = row0 + i + 8 * half;   // C/D layout: VGPR i, lane-half -> M = i + 8*half
      if (row >= MROWS) continue;
      float val = acc[j][i] + bv;
      if (mode == 0) {
        ((__bf16*)out)[(size_t)row * DIM_ + col] = (__bf16)val;
      } else if (mode == 1) {
        int bb  = row / NP1;
        int tok = row - bb * NP1;
        int h = col >> 6, d = col & 63;
        ((__bf16*)out)[((size_t)(bb * H_ + h) * DH + d) * NPAD + tok] = (__bf16)val;
      } else {
        ((float*)out)[(size_t)row * DIM_ + col] = val;
      }
    }
  }
}

// ---- fused attention: scores -> +mask bias -> softmax -> attn out -> attn @ V ----
__global__ __launch_bounds__(32)
void attn_kernel(const __bf16* __restrict__ Qbf, const __bf16* __restrict__ Kbf,
                 const __bf16* __restrict__ Vt, const float* __restrict__ mw,
                 float* __restrict__ attn_out, __bf16* __restrict__ AV) {
  __shared__ float sc[16][NPAD];     // 51.2 KB score/attn tile
  __shared__ float bias_s[NPAD];     // 3.2 KB key bias
  int lane = threadIdx.x;
  int q0 = blockIdx.x * 16;
  int h  = blockIdx.y;
  int b  = blockIdx.z;

  for (int k = lane; k < NPAD; k += 32) {
    float bv;
    if (k < NP1) {
      float fm = (k == 0) ? 1.0f : mw[b * (NP1 - 1) + (k - 1)];
      bv = ALPHA_ * logf(fm + EPS_);
    } else {
      bv = -1e30f;                   // padded keys -> exp() == 0
    }
    bias_s[k] = bv;
  }
  __syncthreads();

  int rowbase = b * NP1;
  int maxrow  = rowbase + NP1 - 1;
  int kq0 = h * DH;
  int n = lane & 15;
  int half = lane >> 4;

  // ---- scores: Q fragments hoisted; K-tile fragments ping-pong buffered ----
  bf16x16 qa0 = load_a_frag(Qbf, DIM_, rowbase + q0, maxrow, kq0,      lane);
  bf16x16 qa1 = load_a_frag(Qbf, DIM_, rowbase + q0, maxrow, kq0 + 32, lane);
  {
    bf16x16 k0A, k1A, k0B, k1B;
    k0A = load_b_frag(Kbf, DIM_, rowbase, maxrow, kq0,      lane);
    k1A = load_b_frag(Kbf, DIM_, rowbase, maxrow, kq0 + 32, lane);
    for (int kt = 0; kt < NPAD / 16; kt += 2) {       // 50 tiles (even)
      int r1 = rowbase + (kt + 1) * 16;
      k0B = load_b_frag(Kbf, DIM_, r1, maxrow, kq0,      lane);
      k1B = load_b_frag(Kbf, DIM_, r1, maxrow, kq0 + 32, lane);
      {
        v8f acc = {};
        acc = wmma_bf16(qa0, k0A, acc);
        acc = wmma_bf16(qa1, k1A, acc);
        int kk = kt * 16 + n;
        float bv = bias_s[kk];
        #pragma unroll
        for (int i = 0; i < 8; ++i)
          sc[i + 8 * half][kk] = acc[i] * 0.125f + bv;   // 1/sqrt(64)
      }
      if (kt + 2 < NPAD / 16) {
        int r2 = rowbase + (kt + 2) * 16;
        k0A = load_b_frag(Kbf, DIM_, r2, maxrow, kq0,      lane);
        k1A = load_b_frag(Kbf, DIM_, r2, maxrow, kq0 + 32, lane);
      }
      {
        v8f acc = {};
        acc = wmma_bf16(qa0, k0B, acc);
        acc = wmma_bf16(qa1, k1B, acc);
        int kk = (kt + 1) * 16 + n;
        float bv = bias_s[kk];
        #pragma unroll
        for (int i = 0; i < 8; ++i)
          sc[i + 8 * half][kk] = acc[i] * 0.125f + bv;
      }
    }
  }
  __syncthreads();

  // ---- softmax: each query row split across a lane pair (lane, lane+16) ----
  {
    int r  = lane & 15;
    int q  = q0 + r;
    int ks = half * (NPAD / 2);
    int ke = ks + (NPAD / 2);
    if (q < NP1) {
      float mx = -1e30f;
      for (int k = ks; k < ke; ++k) mx = fmaxf(mx, sc[r][k]);
      mx = fmaxf(mx, __shfl_xor(mx, 16));
      float sum = 0.0f;
      for (int k = ks; k < ke; ++k) {
        float e = __expf(sc[r][k] - mx);
        sc[r][k] = e;
        sum += e;
      }
      sum += __shfl_xor(sum, 16);
      float inv = 1.0f / sum;
      float* aout = attn_out + ((size_t)(b * H_ + h) * NP1 + q) * NP1;
      for (int k = ks; k < ke; ++k) {
        float v = sc[r][k] * inv;
        sc[r][k] = v;
        if (k < NP1) aout[k] = v;
      }
    } else {
      for (int k = ks; k < ke; ++k) sc[r][k] = 0.0f;  // dead query rows
      (void)__shfl_xor(0.0f, 16);                     // keep shuffle lockstep
      (void)__shfl_xor(0.0f, 16);
    }
  }
  __syncthreads();

  // ---- AV = attn(16x800) @ V(800x64): ping-pong pipelined over 25 k-steps ----
  const __bf16* vbase = Vt + (size_t)(b * H_ + h) * DH * NPAD;
  int m  = lane & 15;
  int kh = lane >> 4;
  v8f acc[4] = {{}, {}, {}, {}};

  bf16x16 avA, avB, vbA[4], vbB[4];
  #pragma unroll
  for (int j = 0; j < 8; ++j) {
    avA[j]     = (__bf16)sc[m][kh * 8 + j];
    avA[8 + j] = (__bf16)sc[m][16 + kh * 8 + j];
  }
  #pragma unroll
  for (int dt = 0; dt < 4; ++dt)
    vbA[dt] = load_b_frag(vbase, NPAD, dt * 16, DH - 1, 0, lane);

  const int KAV = NPAD / 32;                 // 25 (odd)
  for (int kt = 0; kt + 1 < KAV; kt += 2) {  // pairs (0,1)..(22,23)
    int k1 = (kt + 1) * 32;
    #pragma unroll
    for (int j = 0; j < 8; ++j) {
      avB[j]     = (__bf16)sc[m][k1 + kh * 8 + j];
      avB[8 + j] = (__bf16)sc[m][k1 + 16 + kh * 8 + j];
    }
    #pragma unroll
    for (int dt = 0; dt < 4; ++dt)
      vbB[dt] = load_b_frag(vbase, NPAD, dt * 16, DH - 1, k1, lane);
    #pragma unroll
    for (int dt = 0; dt < 4; ++dt)
      acc[dt] = wmma_bf16(avA, vbA[dt], acc[dt]);
    if (kt + 2 < KAV) {
      int k2 = (kt + 2) * 32;
      #pragma unroll
      for (int j = 0; j < 8; ++j) {
        avA[j]     = (__bf16)sc[m][k2 + kh * 8 + j];
        avA[8 + j] = (__bf16)sc[m][k2 + 16 + kh * 8 + j];
      }
      #pragma unroll
      for (int dt = 0; dt < 4; ++dt)
        vbA[dt] = load_b_frag(vbase, NPAD, dt * 16, DH - 1, k2, lane);
    }
    #pragma unroll
    for (int dt = 0; dt < 4; ++dt)
      acc[dt] = wmma_bf16(avB, vbB[dt], acc[dt]);
  }
  // epilogue: final odd k-step (k=24), staged in set A by the last guarded load
  #pragma unroll
  for (int dt = 0; dt < 4; ++dt)
    acc[dt] = wmma_bf16(avA, vbA[dt], acc[dt]);

  #pragma unroll
  for (int dt = 0; dt < 4; ++dt) {
    #pragma unroll
    for (int i = 0; i < 8; ++i) {
      int q = q0 + i + 8 * half;
      if (q < NP1) {
        AV[(size_t)(rowbase + q) * DIM_ + kq0 + dt * 16 + n] = (__bf16)acc[dt][i];
      }
    }
  }
}

extern "C" void kernel_launch(void* const* d_in, const int* in_sizes, int n_in,
                              void* d_out, int out_size, void* d_ws, size_t ws_size,
                              hipStream_t stream) {
  const float* x  = (const float*)d_in[0];
  const float* mw = (const float*)d_in[1];
  const float* Wq = (const float*)d_in[2];
  const float* bq = (const float*)d_in[3];
  const float* Wk = (const float*)d_in[4];
  const float* bk = (const float*)d_in[5];
  const float* Wv = (const float*)d_in[6];
  const float* bv = (const float*)d_in[7];
  const float* Wo = (const float*)d_in[8];
  const float* bo = (const float*)d_in[9];

  char* ws = (char*)d_ws;
  size_t ofs = 0;
  const size_t XB = (size_t)MROWS * DIM_ * sizeof(__bf16);     // 9,646,080
  const size_t WB = (size_t)DIM_ * DIM_ * sizeof(__bf16);      // 1,179,648
  const size_t VB = (size_t)B_ * H_ * DH * NPAD * sizeof(__bf16);
  __bf16* Xbf = (__bf16*)(ws + ofs); ofs += XB;   // reused as AV after projections
  __bf16* Wqb = (__bf16*)(ws + ofs); ofs += WB;
  __bf16* Wkb = (__bf16*)(ws + ofs); ofs += WB;
  __bf16* Wvb = (__bf16*)(ws + ofs); ofs += WB;
  __bf16* Wob = (__bf16*)(ws + ofs); ofs += WB;
  __bf16* Qb  = (__bf16*)(ws + ofs); ofs += XB;
  __bf16* Kb  = (__bf16*)(ws + ofs); ofs += XB;
  __bf16* Vt  = (__bf16*)(ws + ofs); ofs += VB;
  (void)ws_size; (void)in_sizes; (void)n_in; (void)out_size;

  float* outp  = (float*)d_out;
  float* attnp = outp + (size_t)MROWS * DIM_;

  cvt_f32_bf16<<<4096, 256, 0, stream>>>(x,  Xbf, MROWS * DIM_);
  cvt_f32_bf16<<<2048, 256, 0, stream>>>(Wq, Wqb, DIM_ * DIM_);
  cvt_f32_bf16<<<2048, 256, 0, stream>>>(Wk, Wkb, DIM_ * DIM_);
  cvt_f32_bf16<<<2048, 256, 0, stream>>>(Wv, Wvb, DIM_ * DIM_);
  cvt_f32_bf16<<<2048, 256, 0, stream>>>(Wo, Wob, DIM_ * DIM_);

  dim3 gGemm(DIM_ / 64, (MROWS + 15) / 16);   // 12 x 393, 16x64 strip per wave
  proj_gemm<<<gGemm, 32, 0, stream>>>(Xbf, Wqb, bq, Qb, 0);
  proj_gemm<<<gGemm, 32, 0, stream>>>(Xbf, Wkb, bk, Kb, 0);
  proj_gemm<<<gGemm, 32, 0, stream>>>(Xbf, Wvb, bv, Vt, 1);

  attn_kernel<<<dim3(NPAD / 16, H_, B_), 32, 0, stream>>>(Qb, Kb, Vt, mw, attnp, Xbf);

  proj_gemm<<<gGemm, 32, 0, stream>>>(Xbf, Wob, bo, d_out, 2);
}
